// RotaryCrossAttention_9740985827951
// MI455X (gfx1250) — matrix operand order repaired and checked by reading, not compiled, not run on hardware
//
#include <hip/hip_runtime.h>
#include <cmath>

// ---------------------------------------------------------------------------
// RotaryCrossAttention for MI455X (gfx1250, wave32, WMMA).
// Pipeline: LN->f16, W->f16^T, Q/KV projections (WMMA, fused rotary),
// flash attention (WMMA QK^T + PV, online softmax, fused inverse rotary),
// output projection (WMMA + bias) -> f32.
// Global->LDS staging uses GLOBAL_LOAD_ASYNC_TO_LDS_B128 (ASYNCcnt) when the
// toolchain exposes the builtin; V is transposed into LDS so PV fragments are
// contiguous ds_load_b128 like the K path.
// Workspace use: ~74 MB of d_ws for f16 intermediates.
// ---------------------------------------------------------------------------

typedef _Float16 f16;
typedef __attribute__((ext_vector_type(16))) _Float16 v16h;
typedef __attribute__((ext_vector_type(8)))  _Float16 v8h;
typedef __attribute__((ext_vector_type(8)))  float    v8f;
typedef __attribute__((ext_vector_type(4)))  int      v4i;

#define B_     8
#define NQ_    1024
#define NKV_   2048
#define DIM_   512
#define H_     8
#define DH_    64
#define INNER_ 512

#if defined(__has_builtin)
#if __has_builtin(__builtin_amdgcn_global_load_async_to_lds_b128)
#define USE_ASYNC_LDS 1
#endif
#endif

#if defined(USE_ASYNC_LDS)
// Builtin signature (from toolchain diagnostic): (int4 AS1*, int4 AS3*, Ii, Ii)
typedef __attribute__((address_space(1))) v4i* gv4p;
typedef __attribute__((address_space(3))) v4i* lv4p;
// One 16-byte global->LDS async copy per active lane (ASYNCcnt-tracked).
// Integer casts: global flat addr == AS1 addr; low 32 bits of a flat LDS
// address are the LDS byte offset (ISA flat-aperture truncation rule).
__device__ __forceinline__ void async_copy16(void* lds, const void* g) {
  __builtin_amdgcn_global_load_async_to_lds_b128(
      (gv4p)(unsigned long long)g,
      (lv4p)(unsigned)(unsigned long long)lds, 0, 0);
}
#endif

__device__ __forceinline__ void wait_async0() {
#if defined(USE_ASYNC_LDS)
#if __has_builtin(__builtin_amdgcn_s_wait_asynccnt)
  __builtin_amdgcn_s_wait_asynccnt(0);
#else
  asm volatile("s_wait_asynccnt 0x0" ::: "memory");
#endif
#endif
}

// Stage one 16B chunk global->LDS (async when available).
__device__ __forceinline__ void stage16(f16* lds, const f16* g) {
#if defined(USE_ASYNC_LDS)
  async_copy16(lds, g);
#else
  *(v8h*)lds = *(const v8h*)g;
#endif
}

// D = A(16x32 f16) x B(32x16 f16) + C(16x16 f32)
__device__ __forceinline__ v8f wmma32(v16h a, v16h b, v8f c) {
  return __builtin_amdgcn_wmma_f32_16x16x32_f16(false, a, false, b, (short)0, c,
                                                false, false);
}

// A-fragment (16x32, row-major source, stride ld halfs).
// ISA layout: lanes 0-15 (M=lane): VGPR0-3 K=0..7, VGPR4-7 K=16..23;
// lanes 16-31: K=8..15 / 24..31  -> two contiguous 16B loads per lane.
__device__ __forceinline__ v16h load_a_frag(const f16* __restrict__ As, int ld,
                                            int m0, int k0) {
  const int lane = threadIdx.x & 31;
  const int hf = lane >> 4, m = lane & 15;
  const f16* p = As + (m0 + m) * ld + k0 + 8 * hf;
  v8h lo = *(const v8h*)p;
  v8h hi = *(const v8h*)(p + 16);
  v16h a;
#pragma unroll
  for (int i = 0; i < 8; ++i) { a[i] = lo[i]; a[i + 8] = hi[i]; }
  return a;
}

// B-fragment from a [N][K] (pre-transposed) buffer: lanes 0-15 hold K=0..7,16..23,
// lanes 16-31 hold K=8..15,24..31, column = lane&15 -> two contiguous 16B loads.
__device__ __forceinline__ v16h load_bt_frag(const f16* __restrict__ Bt, int ld,
                                             int n0, int k0) {
  const int lane = threadIdx.x & 31;
  const int kh = lane >> 4, n = lane & 15;
  const f16* p = Bt + (n0 + n) * ld + k0 + 8 * kh;
  v8h lo = *(const v8h*)p;
  v8h hi = *(const v8h*)(p + 16);
  v16h b;
#pragma unroll
  for (int i = 0; i < 8; ++i) { b[i] = lo[i]; b[i + 8] = hi[i]; }
  return b;
}

// ---------------------------------------------------------------------------
// LayerNorm (f32 in) -> f16 out; one wave32 per 512-wide row.
// ---------------------------------------------------------------------------
__global__ void __launch_bounds__(128)
ln_f16_kernel(const float* __restrict__ x, const float* __restrict__ g,
              const float* __restrict__ bta, f16* __restrict__ y, int rows) {
  const int lane = threadIdx.x & 31, w = threadIdx.x >> 5;
  const int row = blockIdx.x * 4 + w;
  if (row >= rows) return;
  const float* xr = x + (size_t)row * DIM_;
  float vals[16];
  float s = 0.f;
#pragma unroll
  for (int i = 0; i < 16; ++i) { vals[i] = xr[lane + 32 * i]; s += vals[i]; }
#pragma unroll
  for (int off = 16; off; off >>= 1) s += __shfl_xor(s, off, 32);
  const float mu = s * (1.f / DIM_);
  float v2 = 0.f;
#pragma unroll
  for (int i = 0; i < 16; ++i) { const float d0 = vals[i] - mu; v2 += d0 * d0; }
#pragma unroll
  for (int off = 16; off; off >>= 1) v2 += __shfl_xor(v2, off, 32);
  const float rstd = rsqrtf(v2 * (1.f / DIM_) + 1e-5f);
  f16* yr = y + (size_t)row * DIM_;
#pragma unroll
  for (int i = 0; i < 16; ++i) {
    const int c = lane + 32 * i;
    yr[c] = (f16)((vals[i] - mu) * rstd * g[c] + bta[c]);
  }
}

// ---------------------------------------------------------------------------
// Weight convert+transpose: W[K][N] f32 -> Wt[N][K] f16.
// ---------------------------------------------------------------------------
__global__ void __launch_bounds__(256)
wT_f16_kernel(const float* __restrict__ W, f16* __restrict__ Wt, int K, int N) {
  const int n = blockIdx.x * 32 + (threadIdx.x & 31);
  const int k = blockIdx.y * 8 + (threadIdx.x >> 5);
  if (n < N && k < K) Wt[(size_t)n * K + k] = (f16)W[(size_t)k * N + n];
}

// ---------------------------------------------------------------------------
// GEMM (M x 512) x (512 x N) with WMMA. Block tile 128x64, 8 waves of 32x32.
// MODE 0: Q projection  -> rotary(freqs_q) -> Q16 [b,h,nq,64]
// MODE 1: KV projection -> rotary(freqs_c) -> K16/V16 [b,h,nkv,64]
// MODE 2: output projection -> + bias -> f32 out [b,nq,512]
// ---------------------------------------------------------------------------
template <int MODE>
__global__ void __launch_bounds__(256)
gemm_proj_kernel(const f16* __restrict__ A16, const f16* __restrict__ Bt16,
                 const float* __restrict__ freqs, f16* __restrict__ out0,
                 f16* __restrict__ out1, float* __restrict__ outf,
                 const float* __restrict__ bias, int rowsPerBatch) {
  __shared__ __align__(16) f16 As[128 * 32];
  __shared__ __align__(16) f16 Bs[64 * 32];
  const int t = threadIdx.x;
  const int lane = t & 31, w = t >> 5;
  const int m0 = blockIdx.y * 128;
  const int n0 = blockIdx.x * 64;
  const int mw = (w & 3) * 32, nw = (w >> 2) * 32;

  v8f vzero = {};
  v8f acc[2][2];
#pragma unroll
  for (int i = 0; i < 2; ++i)
#pragma unroll
    for (int j = 0; j < 2; ++j) acc[i][j] = vzero;

  for (int k0 = 0; k0 < 512; k0 += 32) {
    __syncthreads();
    for (int idx = t; idx < 128 * 4; idx += 256) {
      const int r = idx >> 2, c = (idx & 3) * 8;
      stage16(&As[r * 32 + c], &A16[(size_t)(m0 + r) * 512 + k0 + c]);
    }
    for (int idx = t; idx < 64 * 4; idx += 256) {
      const int r = idx >> 2, c = (idx & 3) * 8;
      stage16(&Bs[r * 32 + c], &Bt16[(size_t)(n0 + r) * 512 + k0 + c]);
    }
    if (k0 + 32 < 512) {  // global_prefetch_b8 for next K tile
      __builtin_prefetch(&A16[(size_t)(m0 + (t >> 1)) * 512 + k0 + 32], 0, 1);
      __builtin_prefetch(&Bt16[(size_t)(n0 + (t & 63)) * 512 + k0 + 32], 0, 1);
    }
    wait_async0();
    __syncthreads();
    v16h a0 = load_a_frag(As, 32, mw, 0);
    v16h a1 = load_a_frag(As, 32, mw + 16, 0);
    v16h b0 = load_bt_frag(Bs, 32, nw, 0);
    v16h b1 = load_bt_frag(Bs, 32, nw + 16, 0);
    acc[0][0] = wmma32(a0, b0, acc[0][0]);
    acc[0][1] = wmma32(a0, b1, acc[0][1]);
    acc[1][0] = wmma32(a1, b0, acc[1][0]);
    acc[1][1] = wmma32(a1, b1, acc[1][1]);
  }

  // C layout: VGPR r -> row (r + 8*(lane>>4)), col (lane&15) within 16x16 tile
  const int hf = lane >> 4, nn = lane & 15;
#pragma unroll
  for (int i = 0; i < 2; ++i) {
#pragma unroll
    for (int j = 0; j < 2; ++j) {
#pragma unroll
      for (int r = 0; r < 8; ++r) {
        const int gm = m0 + mw + 16 * i + r + 8 * hf;
        const int gn = n0 + nw + 16 * j + nn;
        const float v = acc[i][j][r];
        const float partner = __shfl_xor(v, 1, 32);  // value at column gn^1
        if (MODE == 2) {
          outf[(size_t)gm * 512 + gn] = v + bias[gn];
        } else {
          const int bb = gm / rowsPerBatch, rr = gm - bb * rowsPerBatch;
          const int cc = gn & 511;         // column within one 512-wide half
          const int hd = cc >> 6, d = cc & 63;
          const float f = freqs[((size_t)bb * rowsPerBatch + rr) * DH_ + d];
          // rotate_half: even d -> -x[d+1], odd d -> +x[d-1]
          const float rot = (d & 1) ? partner : -partner;
          const float res = v * cosf(f) + rot * sinf(f);
          f16* dst = (MODE == 1 && (gn >> 9)) ? out1 : out0;
          dst[(((size_t)bb * H_ + hd) * rowsPerBatch + rr) * DH_ + d] = (f16)res;
        }
      }
    }
  }
}

// ---------------------------------------------------------------------------
// Flash attention: block = (b,h, 128 q rows), 8 waves each own 16 q rows.
// Online softmax entirely wave-local; P reshaped via private LDS strip.
// V staged transposed (Vt[d][ki], stride 72) so PV B-fragments are two
// contiguous ds_load_b128 per fragment, same as the K path.
// Epilogue applies apply_rotary(-freqs_q, O) and merges heads to f16.
// ---------------------------------------------------------------------------
__global__ void __launch_bounds__(256)
attn_kernel(const f16* __restrict__ Q16, const f16* __restrict__ K16,
            const f16* __restrict__ V16, const unsigned char* __restrict__ mask,
            const float* __restrict__ freqs_q, f16* __restrict__ out16) {
  __shared__ __align__(16) f16 Qs[128 * 64];
  __shared__ __align__(16) f16 Ks[64 * 64];
  __shared__ __align__(16) f16 Vt[64 * 72];  // [d][ki], padded stride
  __shared__ __align__(16) f16 Ps[8][16 * 64];

  const int t = threadIdx.x, lane = t & 31, w = t >> 5;
  const int hf = lane >> 4, nn = lane & 15;
  const int q0 = blockIdx.x * 128;
  const int bh = blockIdx.y;
  const int bb = bh >> 3, hd = bh & 7;

  const f16* Qbase = Q16 + ((size_t)bh * NQ_ + q0) * DH_;
  const f16* Kbase = K16 + (size_t)bh * NKV_ * DH_;
  const f16* Vbase = V16 + (size_t)bh * NKV_ * DH_;

  for (int idx = t; idx < 128 * 8; idx += 256) {
    const int r = idx >> 3, c = (idx & 7) * 8;
    stage16(&Qs[r * 64 + c], &Qbase[(size_t)r * 64 + c]);
  }

  v8f vzero = {};
  v8f o[4];
  float m_run[8], l_run[8];
#pragma unroll
  for (int dt = 0; dt < 4; ++dt) o[dt] = vzero;
#pragma unroll
  for (int r = 0; r < 8; ++r) { m_run[r] = -1e30f; l_run[r] = 0.f; }

  wait_async0();
  __syncthreads();
  const v16h qa0 = load_a_frag(Qs, 64, w * 16, 0);
  const v16h qa1 = load_a_frag(Qs, 64, w * 16, 32);

  for (int kv0 = 0; kv0 < NKV_; kv0 += 64) {
    __syncthreads();
    for (int idx = t; idx < 64 * 8; idx += 256) {
      const int r = idx >> 3, c = (idx & 7) * 8;  // r = ki row, c = d base
      stage16(&Ks[r * 64 + c], &Kbase[(size_t)(kv0 + r) * 64 + c]);
      const v8h gv = *(const v8h*)&Vbase[(size_t)(kv0 + r) * 64 + c];
#pragma unroll
      for (int j = 0; j < 8; ++j) Vt[(c + j) * 72 + r] = gv[j];
    }
    if (kv0 + 64 < NKV_) {
      __builtin_prefetch(&Kbase[(size_t)(kv0 + 64 + (t >> 2)) * 64], 0, 1);
      __builtin_prefetch(&Vbase[(size_t)(kv0 + 64 + (t >> 2)) * 64], 0, 1);
    }
    wait_async0();
    __syncthreads();

    // S = Q * K^T (Ks is [ki][d], i.e. already B^T layout for WMMA)
    v8f s[4];
#pragma unroll
    for (int jt = 0; jt < 4; ++jt) {
      v16h kb0 = load_bt_frag(Ks, 64, jt * 16, 0);
      v16h kb1 = load_bt_frag(Ks, 64, jt * 16, 32);
      s[jt] = wmma32(qa1, kb1, wmma32(qa0, kb0, vzero));
    }

    // scale (1/sqrt(64)) + context mask
#pragma unroll
    for (int jt = 0; jt < 4; ++jt) {
      const int kvi = kv0 + jt * 16 + nn;
      const float mk = mask[(size_t)bb * NKV_ + kvi] ? 0.f : -3.0e38f;
#pragma unroll
      for (int r = 0; r < 8; ++r) s[jt][r] = s[jt][r] * 0.125f + mk;
    }

    // online softmax (rows are wave-local; reduce over 16-lane half-groups)
#pragma unroll
    for (int r = 0; r < 8; ++r) {
      float rowmax = s[0][r];
#pragma unroll
      for (int jt = 1; jt < 4; ++jt) rowmax = fmaxf(rowmax, s[jt][r]);
#pragma unroll
      for (int off = 8; off; off >>= 1)
        rowmax = fmaxf(rowmax, __shfl_xor(rowmax, off, 32));
      const float m_new = fmaxf(m_run[r], rowmax);
      const float corr = __expf(m_run[r] - m_new);
      float rowsum = 0.f;
#pragma unroll
      for (int jt = 0; jt < 4; ++jt) {
        const float p = __expf(s[jt][r] - m_new);
        rowsum += p;
        Ps[w][(r + 8 * hf) * 64 + jt * 16 + nn] = (f16)p;
      }
#pragma unroll
      for (int off = 8; off; off >>= 1) rowsum += __shfl_xor(rowsum, off, 32);
      l_run[r] = l_run[r] * corr + rowsum;
      m_run[r] = m_new;
#pragma unroll
      for (int dt = 0; dt < 4; ++dt) o[dt][r] *= corr;
    }

    // make wave's P strip visible to all of its lanes before fragment reads
    __builtin_amdgcn_wave_barrier();
    asm volatile("s_wait_dscnt 0x0" ::: "memory");

    // O += P * V   (Vt is [d][ki] -> contiguous-in-K B fragments)
    const v16h pa0 = load_a_frag(&Ps[w][0], 64, 0, 0);
    const v16h pa1 = load_a_frag(&Ps[w][0], 64, 0, 32);
#pragma unroll
    for (int dt = 0; dt < 4; ++dt) {
      v16h vb0 = load_bt_frag(Vt, 72, dt * 16, 0);
      v16h vb1 = load_bt_frag(Vt, 72, dt * 16, 32);
      o[dt] = wmma32(pa1, vb1, wmma32(pa0, vb0, o[dt]));
    }
  }

  // normalize, inverse rotary (cos f, -sin f), merge heads -> f16
  const int qrow_base = q0 + w * 16;
#pragma unroll
  for (int dt = 0; dt < 4; ++dt) {
#pragma unroll
    for (int r = 0; r < 8; ++r) {
      const int q = qrow_base + r + 8 * hf;
      const int d = dt * 16 + nn;
      const float val = o[dt][r] * (1.f / l_run[r]);
      const float partner = __shfl_xor(val, 1, 32);
      const float rot = (d & 1) ? partner : -partner;
      const float f = freqs_q[((size_t)bb * NQ_ + q) * DH_ + d];
      const float res = val * cosf(f) - rot * sinf(f);
      out16[((size_t)bb * NQ_ + q) * INNER_ + hd * DH_ + d] = (f16)res;
    }
  }
}

// ---------------------------------------------------------------------------
extern "C" void kernel_launch(void* const* d_in, const int* in_sizes, int n_in,
                              void* d_out, int out_size, void* d_ws,
                              size_t ws_size, hipStream_t stream) {
  (void)in_sizes; (void)n_in; (void)out_size; (void)ws_size;
  const float* x_query  = (const float*)d_in[0];
  const float* x_context = (const float*)d_in[1];
  const float* rot_q = (const float*)d_in[2];
  const float* rot_c = (const float*)d_in[3];
  const unsigned char* cmask = (const unsigned char*)d_in[4];  // bool[b,nkv]
  const float* ln_q_g = (const float*)d_in[5];
  const float* ln_q_b = (const float*)d_in[6];
  const float* ln_c_g = (const float*)d_in[7];
  const float* ln_c_b = (const float*)d_in[8];
  const float* Wq  = (const float*)d_in[9];
  const float* Wkv = (const float*)d_in[10];
  const float* Wo  = (const float*)d_in[11];
  const float* bo  = (const float*)d_in[12];
  float* outp = (float*)d_out;

  char* ws = (char*)d_ws;
  size_t off = 0;
  auto wsalloc = [&](size_t bytes) -> void* {
    void* p = ws + off;
    off += (bytes + 255) & ~(size_t)255;
    return p;
  };
  f16* xq16  = (f16*)wsalloc((size_t)B_ * NQ_ * DIM_ * 2);        //  8 MB
  f16* xc16  = (f16*)wsalloc((size_t)B_ * NKV_ * DIM_ * 2);       // 16 MB
  f16* WqT   = (f16*)wsalloc((size_t)INNER_ * DIM_ * 2);          // 0.5 MB
  f16* WkvT  = (f16*)wsalloc((size_t)2 * INNER_ * DIM_ * 2);      //  1 MB
  f16* WoT   = (f16*)wsalloc((size_t)DIM_ * INNER_ * 2);          // 0.5 MB
  f16* Qh    = (f16*)wsalloc((size_t)B_ * H_ * NQ_ * DH_ * 2);    //  8 MB
  f16* Kh    = (f16*)wsalloc((size_t)B_ * H_ * NKV_ * DH_ * 2);   // 16 MB
  f16* Vh    = (f16*)wsalloc((size_t)B_ * H_ * NKV_ * DH_ * 2);   // 16 MB
  f16* att16 = (f16*)wsalloc((size_t)B_ * NQ_ * INNER_ * 2);      //  8 MB

  // 1) LayerNorm -> f16
  ln_f16_kernel<<<(B_ * NQ_) / 4, 128, 0, stream>>>(x_query, ln_q_g, ln_q_b,
                                                    xq16, B_ * NQ_);
  ln_f16_kernel<<<(B_ * NKV_) / 4, 128, 0, stream>>>(x_context, ln_c_g, ln_c_b,
                                                     xc16, B_ * NKV_);
  // 2) Weights -> f16 transposed [N][K]
  wT_f16_kernel<<<dim3(INNER_ / 32, DIM_ / 8), 256, 0, stream>>>(Wq, WqT, DIM_,
                                                                 INNER_);
  wT_f16_kernel<<<dim3((2 * INNER_) / 32, DIM_ / 8), 256, 0, stream>>>(
      Wkv, WkvT, DIM_, 2 * INNER_);
  wT_f16_kernel<<<dim3(DIM_ / 32, INNER_ / 8), 256, 0, stream>>>(Wo, WoT,
                                                                 INNER_, DIM_);
  // 3) Q projection + rotary
  gemm_proj_kernel<0><<<dim3(INNER_ / 64, (B_ * NQ_) / 128), 256, 0, stream>>>(
      xq16, WqT, rot_q, Qh, nullptr, nullptr, nullptr, NQ_);
  // 4) KV projection + rotary (reference applies rotary to V as well)
  gemm_proj_kernel<1><<<dim3((2 * INNER_) / 64, (B_ * NKV_) / 128), 256, 0,
                        stream>>>(xc16, WkvT, rot_c, Kh, Vh, nullptr, nullptr,
                                  NKV_);
  // 5) flash attention + inverse rotary + head merge
  attn_kernel<<<dim3(NQ_ / 128, B_ * H_), 256, 0, stream>>>(Qh, Kh, Vh, cmask,
                                                            rot_q, att16);
  // 6) output projection + bias -> f32
  gemm_proj_kernel<2><<<dim3(DIM_ / 64, (B_ * NQ_) / 128), 256, 0, stream>>>(
      att16, WoT, nullptr, nullptr, nullptr, outp, bo, NQ_);
}
